// CustomMultiHeadAttention_11596411699912
// MI455X (gfx1250) — matrix operand order repaired and checked by reading, not compiled, not run on hardware
//
#include <hip/hip_runtime.h>
#include <hip/hip_bf16.h>

// ---------- types ----------
typedef __bf16 bf16;
typedef __bf16 v16bf __attribute__((ext_vector_type(16)));
typedef float  v8f   __attribute__((ext_vector_type(8)));
typedef unsigned int u32;
typedef unsigned int u32x4 __attribute__((ext_vector_type(4)));
typedef float f32x4 __attribute__((ext_vector_type(4)));

union F16x16 { v16bf v; u32x4 q[2]; };  // 32 bytes: one WMMA bf16 A/B fragment

__device__ __forceinline__ u32 bfbits(float x) {        // RNE fp32 -> bf16 bits
    u32 u = __builtin_bit_cast(u32, x);
    return (u + 0x7FFFu + ((u >> 16) & 1u)) >> 16;
}
__device__ __forceinline__ bf16 f2bf(float x) {
    unsigned short h = (unsigned short)bfbits(x);
    return __builtin_bit_cast(bf16, h);
}
__device__ __forceinline__ u32 pk2(float a, float b) {  // two bf16 packed in u32
    return bfbits(a) | (bfbits(b) << 16);
}

__device__ __forceinline__ v8f wmma_bf16(v16bf a, v16bf b, v8f c) {
    // D = A(16x32 bf16) x B(32x16 bf16) + C(16x16 f32)
    return __builtin_amdgcn_wmma_f32_16x16x32_bf16(
        false, a, false, b, (short)0, c, false, false);
}

// ====================================================================
// One-time fp32 -> bf16 conversion (8 elements / thread, 16B stores)
// ====================================================================
__global__ __launch_bounds__(256) void cvt_bf16(
    const float* __restrict__ x, bf16* __restrict__ y, int n8)
{
    int i = blockIdx.x * blockDim.x + threadIdx.x;
    if (i >= n8) return;
    const f32x4* p = (const f32x4*)x + (size_t)i * 2;
    f32x4 a = p[0], c = p[1];
    u32x4 o;
    o[0] = pk2(a[0], a[1]); o[1] = pk2(a[2], a[3]);
    o[2] = pk2(c[0], c[1]); o[3] = pk2(c[2], c[3]);
    ((u32x4*)y)[i] = o;
}

// ====================================================================
// GEMM:  Y = (X @ W^T + bias) * scale    (X, W already bf16)
//   X: M x K bf16 row-major, W: N x K bf16 row-major (torch Linear weight)
//   one wave -> 32(M) x 64(N) tile, K-loop step 32, 8 WMMAs / step.
// mode 0: Q -> bf16 [B,H,T,HD]   (scale = HD^-0.5)
// mode 1: K -> bf16 [B,H,S,HD]
// mode 2: V -> bf16 [B,H,HD,S]   (transposed for P@V B-fragments)
// mode 3: O -> fp32 row-major [M,N]  (= [T,B,C] flat)
// ====================================================================
__global__ __launch_bounds__(128) void gemm_bf16(
    const bf16* __restrict__ X, const bf16* __restrict__ W,
    const float* __restrict__ bias, void* __restrict__ Out,
    int M, int Kdim, int N, float scale, int mode,
    int Bsz, int Lseq, int Hn, int Hd)
{
    const int lane = threadIdx.x & 31;
    const int wave = (blockIdx.x * blockDim.x + threadIdx.x) >> 5;
    const int mtiles = M >> 5;           // 32-row M tiles
    const int mt = wave % mtiles;
    const int nt = wave / mtiles;        // 64-wide N tile
    if (nt >= (N >> 6)) return;
    const int m0 = mt << 5;
    const int n0 = nt << 6;
    const int hi = lane >> 4;            // half-wave id
    const int lo = lane & 15;

    v8f acc[2][4] = {};

    const bf16* xrow0 = X + (size_t)(m0 + lo) * Kdim;        // A rows: M = lo
    const bf16* xrow1 = xrow0 + (size_t)16 * Kdim;           // M = 16 + lo
    const int aklo = hi * 8;             // A-frag K offset per half-wave

    for (int k0 = 0; k0 < Kdim; k0 += 32) {
        __builtin_prefetch(xrow0 + k0 + 512, 0, 3);
        __builtin_prefetch(xrow1 + k0 + 512, 0, 3);
        // ---- A fragments (16x32 bf16 each) ----
        F16x16 a0, a1;
        {
            const u32x4* p0 = (const u32x4*)(xrow0 + k0 + aklo);
            a0.q[0] = p0[0];             // K = k0+aklo+0..7
            a0.q[1] = p0[2];             // K = k0+aklo+16..23
            const u32x4* p1 = (const u32x4*)(xrow1 + k0 + aklo);
            a1.q[0] = p1[0];
            a1.q[1] = p1[2];
        }
        // ---- 4 B fragments (32x16 bf16): lane = col n, K = hi*8+0..7, 16+hi*8+0..7 ----
        #pragma unroll
        for (int j = 0; j < 4; ++j) {
            const int n = n0 + j * 16 + lo;
            const u32x4* wp = (const u32x4*)(W + (size_t)n * Kdim + k0 + hi * 8);
            F16x16 b;
            b.q[0] = wp[0]; b.q[1] = wp[2];
            acc[0][j] = wmma_bf16(a0.v, b.v, acc[0][j]);
            acc[1][j] = wmma_bf16(a1.v, b.v, acc[1][j]);
        }
    }

    // ---- epilogue: bias, scale, layout-specific store ----
    #pragma unroll
    for (int mi = 0; mi < 2; ++mi) {
        #pragma unroll
        for (int j = 0; j < 4; ++j) {
            const int n = n0 + j * 16 + lo;
            const float bv = bias[n];
            #pragma unroll
            for (int r = 0; r < 8; ++r) {
                const float val = (acc[mi][j][r] + bv) * scale;  // row r+8*hi
                const int mm = m0 + mi * 16 + r + 8 * hi;
                if (mode == 3) {
                    ((float*)Out)[(size_t)mm * N + n] = val;
                } else {
                    const int t = mm / Bsz, b = mm % Bsz;        // row = l*B + b
                    const int h = n / Hd, hd = n % Hd;
                    size_t idx;
                    if (mode == 2)
                        idx = (((size_t)(b * Hn + h)) * Hd + hd) * Lseq + t;  // V^T
                    else
                        idx = (((size_t)(b * Hn + h)) * Lseq + t) * Hd + hd;
                    ((bf16*)Out)[idx] = f2bf(val);
                }
            }
        }
    }
}

// ====================================================================
// Flash attention: one wave = one (b,h) and a 16-row T tile.
//   Q [BH][T][64] bf16, K [BH][S][64] bf16, Vt [BH][64][S] bf16
//   online softmax in fp32, P re-shaped D-layout -> A-layout via LDS.
//   Output: bf16 row-major [T*B][C]  (A layout for the final projection)
// ====================================================================
__global__ __launch_bounds__(128) void attn_kernel(
    const bf16* __restrict__ Q, const bf16* __restrict__ Kb,
    const bf16* __restrict__ Vt, const unsigned char* __restrict__ mask,
    bf16* __restrict__ Ao)
{
    const int T = 1024, S = 1024, HD = 64, H = 16, Bsz = 8, C = 1024;
    const int lane = threadIdx.x & 31;
    const int wid  = threadIdx.x >> 5;
    const int wave = blockIdx.x * 4 + wid;
    const int ttiles = T / 16;
    const int tt = wave % ttiles;
    const int bh = wave / ttiles;
    const int b  = bh / H;
    const int t0 = tt * 16;
    const int hi = lane >> 4;
    const int lo = lane & 15;

    __shared__ __align__(16) bf16 psh[4][16 * 32];
    bf16* pp = psh[wid];

    // resident Q A-fragments (16 x 64 = two 16x32 frags)
    F16x16 qa[2];
    {
        const bf16* qrow = Q + ((size_t)bh * T + (t0 + lo)) * HD + hi * 8;
        #pragma unroll
        for (int f = 0; f < 2; ++f) {
            const u32x4* p = (const u32x4*)(qrow + f * 32);
            qa[f].q[0] = p[0];   // K = f*32 + hi*8 + 0..7
            qa[f].q[1] = p[2];   // K = f*32 + hi*8 + 16..23
        }
    }

    v8f o[4] = {};
    float mrow[8], lrow[8];
    #pragma unroll
    for (int r = 0; r < 8; ++r) { mrow[r] = -3.0e38f; lrow[r] = 0.0f; }

    const bf16* kbh = Kb + (size_t)bh * S * HD;
    const bf16* vbh = Vt + (size_t)bh * HD * S;

    for (int s0 = 0; s0 < S; s0 += 32) {
        // ---- scores: 16 x 32 tile = two 16x16 C frags ----
        v8f c0 = {}, c1 = {};
        #pragma unroll
        for (int f = 0; f < 2; ++f) {
            F16x16 kb;
            const u32x4* p = (const u32x4*)(kbh + (size_t)(s0 + lo) * HD + f * 32 + hi * 8);
            kb.q[0] = p[0]; kb.q[1] = p[2];
            c0 = wmma_bf16(qa[f].v, kb.v, c0);
        }
        #pragma unroll
        for (int f = 0; f < 2; ++f) {
            F16x16 kb;
            const u32x4* p = (const u32x4*)(kbh + (size_t)(s0 + 16 + lo) * HD + f * 32 + hi * 8);
            kb.q[0] = p[0]; kb.q[1] = p[2];
            c1 = wmma_bf16(qa[f].v, kb.v, c1);
        }

        // ---- key padding mask (this lane's columns are lo / lo+16) ----
        if (mask[(size_t)b * S + s0 + lo]) {
            #pragma unroll
            for (int r = 0; r < 8; ++r) c0[r] = -INFINITY;
        }
        if (mask[(size_t)b * S + s0 + 16 + lo]) {
            #pragma unroll
            for (int r = 0; r < 8; ++r) c1[r] = -INFINITY;
        }

        // ---- online softmax (row = r + 8*hi; reduce across 16-lane half-wave) ----
        float alpha[8];
        #pragma unroll
        for (int r = 0; r < 8; ++r) {
            float mx = fmaxf(c0[r], c1[r]);
            #pragma unroll
            for (int off = 1; off < 16; off <<= 1)
                mx = fmaxf(mx, __shfl_xor(mx, off, 32));
            const float mn = fmaxf(mrow[r], mx);
            alpha[r] = __expf(mrow[r] - mn);
            mrow[r] = mn;
            const float p0 = __expf(c0[r] - mn);
            const float p1 = __expf(c1[r] - mn);
            c0[r] = p0; c1[r] = p1;
            float rs = p0 + p1;
            #pragma unroll
            for (int off = 1; off < 16; off <<= 1)
                rs += __shfl_xor(rs, off, 32);
            lrow[r] = lrow[r] * alpha[r] + rs;
        }
        #pragma unroll
        for (int r = 0; r < 8; ++r) {
            o[0][r] *= alpha[r]; o[1][r] *= alpha[r];
            o[2][r] *= alpha[r]; o[3][r] *= alpha[r];
        }

        // ---- P: D-layout -> bf16 LDS -> A-layout fragment ----
        __syncthreads();   // WAR vs previous iteration's loads
        #pragma unroll
        for (int r = 0; r < 8; ++r) {
            pp[(r + 8 * hi) * 32 + lo]      = f2bf(c0[r]);
            pp[(r + 8 * hi) * 32 + 16 + lo] = f2bf(c1[r]);
        }
        __syncthreads();   // RAW
        F16x16 pa;
        {
            const u32x4* p = (const u32x4*)(pp + lo * 32 + hi * 8);
            pa.q[0] = p[0];   // cols hi*8 + 0..7
            pa.q[1] = p[2];   // cols hi*8 + 16..23
        }

        // ---- O += P(16x32) @ V(32x64): B frags from V^T, contiguous in s ----
        #pragma unroll
        for (int j = 0; j < 4; ++j) {
            F16x16 vb;
            const u32x4* p = (const u32x4*)(vbh + (size_t)(j * 16 + lo) * S + s0 + hi * 8);
            vb.q[0] = p[0]; vb.q[1] = p[2];
            o[j] = wmma_bf16(pa.v, vb.v, o[j]);
        }
    }

    // ---- normalize and store bf16 row-major [T*B][C] ----
    #pragma unroll
    for (int r = 0; r < 8; ++r) {
        const float inv = 1.0f / lrow[r];
        o[0][r] *= inv; o[1][r] *= inv; o[2][r] *= inv; o[3][r] *= inv;
    }
    const int h = bh % H;
    #pragma unroll
    for (int j = 0; j < 4; ++j) {
        #pragma unroll
        for (int r = 0; r < 8; ++r) {
            const int t = t0 + r + 8 * hi;
            const int n = h * 64 + j * 16 + lo;
            Ao[((size_t)t * Bsz + b) * C + n] = f2bf(o[j][r]);
        }
    }
}

// ====================================================================
extern "C" void kernel_launch(void* const* d_in, const int* in_sizes, int n_in,
                              void* d_out, int out_size, void* d_ws, size_t ws_size,
                              hipStream_t stream)
{
    const float* query = (const float*)d_in[0];
    const float* key_  = (const float*)d_in[1];
    const float* value = (const float*)d_in[2];
    const unsigned char* mask = (const unsigned char*)d_in[3];
    const float* Wq = (const float*)d_in[4];
    const float* bq = (const float*)d_in[5];
    const float* Wk = (const float*)d_in[6];
    const float* bk = (const float*)d_in[7];
    const float* Wv = (const float*)d_in[8];
    const float* bv = (const float*)d_in[9];
    const float* Wo = (const float*)d_in[10];
    const float* bo = (const float*)d_in[11];

    const int T = 1024, S = 1024, Bsz = 8, C = 1024, H = 16, HD = 64;
    const float SCALE = 0.125f;                    // 64^-0.5

    // ---- workspace carve-up (all bf16 except none) ----
    const size_t act = (size_t)T * Bsz * C;        // 8.4M elements
    const size_t wel = (size_t)C * C;              // 1M elements
    bf16* Xq  = (bf16*)d_ws;                       // bf16 copies of inputs
    bf16* Xk  = Xq + act;
    bf16* Xv  = Xk + act;
    bf16* Wqb = Xv + act;                          // bf16 weights
    bf16* Wkb = Wqb + wel;
    bf16* Wvb = Wkb + wel;
    bf16* Wob = Wvb + wel;
    bf16* Qb  = Wob + wel;                         // projected tensors
    bf16* Kbf = Qb + act;
    bf16* Vt  = Kbf + act;
    bf16* Ao  = Vt + act;                          // attention output

    // ---- one-time fp32 -> bf16 conversions ----
    {
        const int a8 = (int)(act / 8), w8 = (int)(wel / 8);
        cvt_bf16<<<dim3((a8 + 255) / 256), dim3(256), 0, stream>>>(query, Xq, a8);
        cvt_bf16<<<dim3((a8 + 255) / 256), dim3(256), 0, stream>>>(key_,  Xk, a8);
        cvt_bf16<<<dim3((a8 + 255) / 256), dim3(256), 0, stream>>>(value, Xv, a8);
        cvt_bf16<<<dim3((w8 + 255) / 256), dim3(256), 0, stream>>>(Wq, Wqb, w8);
        cvt_bf16<<<dim3((w8 + 255) / 256), dim3(256), 0, stream>>>(Wk, Wkb, w8);
        cvt_bf16<<<dim3((w8 + 255) / 256), dim3(256), 0, stream>>>(Wv, Wvb, w8);
        cvt_bf16<<<dim3((w8 + 255) / 256), dim3(256), 0, stream>>>(Wo, Wob, w8);
    }

    const int M = T * Bsz;                         // 8192
    const int waves = (M / 32) * (C / 64);         // 4096 waves
    dim3 blk(128), grd(waves / 4);

    gemm_bf16<<<grd, blk, 0, stream>>>(Xq, Wqb, bq, Qb,  M, C, C, SCALE, 0, Bsz, T, H, HD);
    gemm_bf16<<<grd, blk, 0, stream>>>(Xk, Wkb, bk, Kbf, M, C, C, 1.0f,  1, Bsz, S, H, HD);
    gemm_bf16<<<grd, blk, 0, stream>>>(Xv, Wvb, bv, Vt,  M, C, C, 1.0f,  2, Bsz, S, H, HD);

    const int awaves = Bsz * H * (T / 16);         // 8192 waves
    attn_kernel<<<dim3(awaves / 4), blk, 0, stream>>>(Qb, Kbf, Vt, mask, Ao);

    gemm_bf16<<<grd, blk, 0, stream>>>(Ao, Wob, bo, d_out, M, C, C, 1.0f, 3, Bsz, T, H, HD);
}
